// StructuralSparseBlock_76673756168878
// MI455X (gfx1250) — matrix operand order repaired and checked by reading, not compiled, not run on hardware
//
#include <hip/hip_runtime.h>
#include <hip/hip_bf16.h>

typedef __attribute__((ext_vector_type(16))) _Float16 v16h;
typedef __attribute__((ext_vector_type(8)))  _Float16 v8h;
typedef __attribute__((ext_vector_type(8)))  float    v8f;
typedef __attribute__((ext_vector_type(4)))  float    v4f;

#define BATCH   16384
#define EMB     2048
#define GROUPS  32
#define DMID    128
#define NLAYER  2
#define BM      128     // batch rows per workgroup
#define AK      136     // activation buffer row stride in halfs (bank pad, 16B align)
#define WK      136     // w_lds row stride in halfs
#define GS2     132     // f32 h+g buffer row stride (aliases w_lds, 132*4*128 <= 69632)

#define WT_HALFS_PER_GROUP (NLAYER * DMID * WK)          // 34816 halfs = 69632 B
#define WT_BYTES ((size_t)GROUPS * WT_HALFS_PER_GROUP * 2)

union V16U { v16h v; v8h h[2]; };

// Branchless tanh-form GELU, constants folded into the exponent:
//   gelu(x) ~= x * rcp(1 + exp2(x * (c1 + c3*x^2)))
//   c1 = -2*log2(e)*sqrt(2/pi), c3 = c1*0.044715
// 5 VALU + v_exp_f32 + v_rcp_f32, no branches.
__device__ __forceinline__ float gelu_fast(float x) {
    const float x2 = x * x;
    const float t  = __builtin_fmaf(x2, -0.1029432372f, -2.3022081964f);
    const float z  = __builtin_amdgcn_exp2f(x * t);
    return x * __builtin_amdgcn_rcpf(1.0f + z);
}

// ---- pre-pass: W [G,L,D,D] f32 row-major -> wt [G*L][n][WK] f16, transposed
// and pre-padded so the main kernel can DMA it straight into LDS.
__global__ __launch_bounds__(256)
void wt_prep_kernel(const float* __restrict__ W, _Float16* __restrict__ wt) {
    const int i  = blockIdx.x * 256 + threadIdx.x;   // G*L*D*D = 1,048,576 total
    const int n  = i & (DMID - 1);
    const int k  = (i >> 7) & (DMID - 1);
    const int gl = i >> 14;
    wt[((size_t)gl * DMID + n) * WK + k] =
        (_Float16)W[((size_t)gl * DMID + k) * DMID + n];
}

template<bool USE_WS>
__global__ __launch_bounds__(256)
void ssblock_76673756168878_kernel(const float* __restrict__ x,
                                   const int*   __restrict__ indices,
                                   const float* __restrict__ W,
                                   const _Float16* __restrict__ wt,
                                   const float* __restrict__ bias,
                                   const float* __restrict__ gamma,
                                   const float* __restrict__ beta,
                                   float* __restrict__ out)
{
    // 138.5 KB total -> 2 workgroups resident per WGP (320 KB limit)
    __shared__ alignas(16) _Float16 w_lds[NLAYER][DMID * WK]; // W^T f16 [l][n][k];
                                                              // reused as f32 h+g after layer 1
    __shared__ alignas(16) _Float16 buf0[BM * AK];            // gathered g (residual source)
    __shared__ alignas(16) _Float16 buf1[BM * AK];            // h1 = gelu(g W0 + b0)
    __shared__ float biasS[NLAYER][DMID];
    __shared__ float gammaS[DMID];
    __shared__ float betaS[DMID];
    __shared__ int   idxS[DMID];

    const int tid     = threadIdx.x;
    const int g       = blockIdx.x & (GROUPS - 1);   // group fastest -> L2 reuse of x
    const int tile    = blockIdx.x >> 5;
    const int rowBase = tile * BM;

    // ---- stage 0: per-group parameters ----
    if (tid < DMID) {
        idxS[tid]     = indices[g * DMID + tid];
        biasS[0][tid] = bias[(g * NLAYER + 0) * DMID + tid];
        biasS[1][tid] = bias[(g * NLAYER + 1) * DMID + tid];
        gammaS[tid]   = gamma[g * DMID + tid];
        betaS[tid]    = beta[g * DMID + tid];
    }

    // ---- weights into LDS ----
    if constexpr (USE_WS) {
        // Async DMA of the pre-transposed, pre-padded f16 weight image:
        // 69632 B = 4352 x 16B chunks, 17 per thread, tracked by ASYNCcnt.
        // Overlaps with the gather phase below.
        const char* wsrc = (const char*)(wt + (size_t)g * WT_HALFS_PER_GROUP);
        const uint32_t lds0 = (uint32_t)(uintptr_t)(&w_lds[0][0]);
        for (int i = tid; i < WT_HALFS_PER_GROUP / 8; i += 256) {
            const uint32_t loff = lds0 + (uint32_t)i * 16u;
            const uint32_t goff = (uint32_t)i * 16u;
            asm volatile("global_load_async_to_lds_b128 %0, %1, %2"
                         :: "v"(loff), "v"(goff), "s"(wsrc) : "memory");
        }
    } else {
        // Fallback: transpose + convert in-kernel (each thread owns column n).
        const int n    = tid & (DMID - 1);
        const int kOff = tid >> 7;
        const float* Wg = W + (size_t)g * NLAYER * DMID * DMID;
        #pragma unroll
        for (int l = 0; l < NLAYER; ++l) {
            #pragma unroll 4
            for (int jj = 0; jj < DMID / 2; ++jj) {
                const int k = kOff + 2 * jj;
                w_lds[l][n * WK + k] =
                    (_Float16)Wg[((size_t)l * DMID + k) * DMID + n];
            }
        }
    }
    __syncthreads();  // (1) idxS ready before gather

    // ---- stage 1: gather. Each thread owns one column d (constant),
    //      walks 64 rows with stride 2; single f16 store per element. ----
    {
        const int d    = tid & (DMID - 1);
        const int rOff = tid >> 7;                   // 0 or 1
        const int col  = idxS[d];
        const float* xp = x + (size_t)rowBase * EMB + col;
        #pragma unroll 8
        for (int j = 0; j < BM / 2; ++j) {
            const int r = rOff + 2 * j;
            buf0[r * AK + d] = (_Float16)xp[(size_t)r * EMB];
        }
    }
    if constexpr (USE_WS) {
        asm volatile("s_wait_asynccnt 0x0" ::: "memory");  // weight DMA done
    }
    __syncthreads();  // (2) gather + weights visible to all waves

    const int lane   = tid & 31;
    const int wave   = tid >> 5;
    const int laneLo = lane & 15;
    const int laneHi = lane >> 4;
    const int M0     = wave * 16;     // each wave owns a 16-row strip
    const int mA     = M0 + laneLo;   // A-fragment row for this lane

    // ================= layer 0: h1 = gelu(g @ W0 + b0), buf0 -> buf1 =========
    {
        // A fragments: 16x32 f16 per k-tile. ISA layout: lane = M (0..15),
        // laneHi = h selects 8-K block; VGPRs 0-3 = K 8h..8h+7, VGPRs 4-7 = K 16+8h..16+8h+7.
        V16U va[4];
        #pragma unroll
        for (int kt = 0; kt < 4; ++kt) {
            const int k0 = kt * 32 + laneHi * 8;
            va[kt].h[0] = *reinterpret_cast<const v8h*>(&buf0[mA * AK + k0]);
            va[kt].h[1] = *reinterpret_cast<const v8h*>(&buf0[mA * AK + k0 + 16]);
        }
        const float* bl = biasS[0];
        #pragma unroll
        for (int nt = 0; nt < 8; ++nt) {
            v8f acc = {};
            const int n = nt * 16 + laneLo;
            #pragma unroll
            for (int kt = 0; kt < 4; ++kt) {
                // B fragment 32x16: lane = column n, laneHi selects 8-K block,
                // K = 8h..8h+7 then 16+8h..16+8h+7 (matches A) -> two ds_load_b128
                V16U vb;
                const int kB = kt * 32 + laneHi * 8;
                vb.h[0] = *reinterpret_cast<const v8h*>(&w_lds[0][n * WK + kB]);
                vb.h[1] = *reinterpret_cast<const v8h*>(&w_lds[0][n * WK + kB + 16]);
                acc = __builtin_amdgcn_wmma_f32_16x16x32_f16(
                        false, va[kt].v, false, vb.v, (short)0, acc, false, false);
            }
            // C/D layout: lane holds column n; VGPR r holds row M0 + 8*laneHi + r
            const float bn = bl[n];
            #pragma unroll
            for (int r = 0; r < 8; ++r) {
                const int m = M0 + laneHi * 8 + r;
                buf1[m * AK + n] = (_Float16)gelu_fast(acc[r] + bn);
            }
        }
    }
    __syncthreads();  // (3) h1 complete (also orders buf1 writes vs reads below)

    // ========= layer 1: h2 = gelu(h1 @ W1 + b1); h = h2 + g; store f32 =======
    {
        V16U va[4];
        #pragma unroll
        for (int kt = 0; kt < 4; ++kt) {
            const int k0 = kt * 32 + laneHi * 8;
            va[kt].h[0] = *reinterpret_cast<const v8h*>(&buf1[mA * AK + k0]);
            va[kt].h[1] = *reinterpret_cast<const v8h*>(&buf1[mA * AK + k0 + 16]);
        }
        // Compute all 8 accumulator tiles first: 8 independent 4-WMMA chains,
        // and all w_lds B reads complete before we alias it as the f32 buffer.
        v8f acc[8];
        #pragma unroll
        for (int nt = 0; nt < 8; ++nt) {
            acc[nt] = {};
            const int n = nt * 16 + laneLo;
            #pragma unroll
            for (int kt = 0; kt < 4; ++kt) {
                V16U vb;
                const int kB = kt * 32 + laneHi * 8;
                vb.h[0] = *reinterpret_cast<const v8h*>(&w_lds[1][n * WK + kB]);
                vb.h[1] = *reinterpret_cast<const v8h*>(&w_lds[1][n * WK + kB + 16]);
                acc[nt] = __builtin_amdgcn_wmma_f32_16x16x32_f16(
                        false, va[kt].v, false, vb.v, (short)0, acc[nt], false, false);
            }
        }
        __syncthreads();  // (4) all waves done reading w_lds -> safe to alias

        float* hBuf = reinterpret_cast<float*>(&w_lds[0][0]);  // [BM][GS2] f32
        const float* bl = biasS[1];
        #pragma unroll
        for (int nt = 0; nt < 8; ++nt) {
            const int n = nt * 16 + laneLo;
            const float bn = bl[n];
            #pragma unroll
            for (int r = 0; r < 8; ++r) {
                const int m = M0 + laneHi * 8 + r;
                const float y = gelu_fast(acc[nt][r] + bn);
                hBuf[m * GS2 + n] = y + (float)buf0[m * AK + n];  // residual
            }
        }
    }

    // ---- stage 3: per-row LayerNorm (eps=1e-3) over own strip, write output.
    // Rows of this wave's strip were written only by this wave (in-order LDS).
    {
        const float* hBuf = reinterpret_cast<const float*>(&w_lds[0][0]);
        const int m  = M0 + laneLo;   // 2 lanes per row (laneHi = half selector)
        const int c0 = laneHi * 64;
        const float* rowp = &hBuf[m * GS2];
        float s = 0.f, s2 = 0.f;
        #pragma unroll
        for (int j = 0; j < 64; j += 4) {
            const v4f h4 = *reinterpret_cast<const v4f*>(rowp + c0 + j);
            s  += h4.x + h4.y + h4.z + h4.w;
            s2 += h4.x*h4.x + h4.y*h4.y + h4.z*h4.z + h4.w*h4.w;
        }
        s  += __shfl_xor(s, 16, 32);
        s2 += __shfl_xor(s2, 16, 32);
        const float mean = s * (1.0f / (float)DMID);
        const float var  = s2 * (1.0f / (float)DMID) - mean * mean;
        const float rstd = __builtin_amdgcn_rsqf(var + 1e-3f);
        float* op = out + ((size_t)(rowBase + m) * GROUPS + g) * DMID + c0;
        #pragma unroll
        for (int j = 0; j < 64; j += 4) {
            const v4f h4 = *reinterpret_cast<const v4f*>(rowp + c0 + j);
            const v4f g4 = *reinterpret_cast<const v4f*>(&gammaS[c0 + j]);
            const v4f b4 = *reinterpret_cast<const v4f*>(&betaS[c0 + j]);
            v4f o;
            o.x = (h4.x - mean) * rstd * g4.x + b4.x;
            o.y = (h4.y - mean) * rstd * g4.y + b4.y;
            o.z = (h4.z - mean) * rstd * g4.z + b4.z;
            o.w = (h4.w - mean) * rstd * g4.w + b4.w;
            *reinterpret_cast<v4f*>(op + j) = o;
        }
    }
}

extern "C" void kernel_launch(void* const* d_in, const int* in_sizes, int n_in,
                              void* d_out, int out_size, void* d_ws, size_t ws_size,
                              hipStream_t stream) {
    (void)in_sizes; (void)n_in; (void)out_size;
    const float* x       = (const float*)d_in[0];
    const int*   indices = (const int*)  d_in[1];
    const float* W       = (const float*)d_in[2];
    const float* b       = (const float*)d_in[3];
    const float* gamma   = (const float*)d_in[4];
    const float* beta    = (const float*)d_in[5];
    float* out = (float*)d_out;

    const int nBlocks = GROUPS * (BATCH / BM);  // 32 * 128 = 4096

    if (d_ws != nullptr && ws_size >= WT_BYTES) {
        _Float16* wt = (_Float16*)d_ws;
        const int prepBlocks = (GROUPS * NLAYER * DMID * DMID) / 256;  // 4096
        wt_prep_kernel<<<dim3(prepBlocks), dim3(256), 0, stream>>>(W, wt);
        ssblock_76673756168878_kernel<true><<<dim3(nBlocks), dim3(256), 0, stream>>>(
            x, indices, W, wt, b, gamma, beta, out);
    } else {
        ssblock_76673756168878_kernel<false><<<dim3(nBlocks), dim3(256), 0, stream>>>(
            x, indices, W, nullptr, b, gamma, beta, out);
    }
}
